// PatchCrossAttn_13168369729763
// MI455X (gfx1250) — compile-verified, matrix operand order checked by reading, not run on hardware
//
#include <hip/hip_runtime.h>
#include <hip/hip_bf16.h>
#include <stdint.h>

typedef __attribute__((ext_vector_type(16))) __bf16 v16bf;
typedef __attribute__((ext_vector_type(8)))  float  v8f;
typedef __attribute__((ext_vector_type(4)))  unsigned int v4u;
typedef __attribute__((ext_vector_type(8)))  int v8i;
typedef __attribute__((ext_vector_type(4)))  int v4i;

#ifndef __has_builtin
#define __has_builtin(x) 0
#endif
#if defined(__HIP_DEVICE_COMPILE__) && __has_builtin(__builtin_amdgcn_tensor_load_to_lds) && __has_builtin(__builtin_amdgcn_s_wait_tensorcnt)
#define HAVE_TDM 1
#else
#define HAVE_TDM 0
#endif

union Frag { v4u q[2]; unsigned int u[8]; v16bf v; };

// TDM group1 word0 configs: workgroup_mask=0 | data_size | pad_enable | pad_interval | pad_amount
// bf16 rows: 32 halfs (16 dwords) data + 4 dwords pad  -> row stride 40 halfs (SA)
#define TDW_BF16 ((1u << 16) | (1u << 20) | (3u << 22) | (3u << 25))
// f32 rows: 32 dwords data + 1 dword pad               -> row stride 33 floats
#define TDW_F32  ((2u << 16) | (1u << 20) | (4u << 22) | (0u << 25))

#if HAVE_TDM
__device__ __forceinline__ void tdm_load_2d(unsigned int lds_addr, unsigned long long gaddr,
                                            unsigned int w0cfg, unsigned int td0, unsigned int td1,
                                            unsigned int tile0, unsigned int tile1, unsigned int stride0)
{
    v4u g0 = { 1u,                                   // count=1, user descriptor, no gather
               lds_addr,                             // LDS byte address
               (unsigned int)gaddr,                  // global addr [31:0]
               (unsigned int)((gaddr >> 32) & 0x01FFFFFFull) | (2u << 30) };  // addr[56:32] | type=2
    v8i g1 = { (int)w0cfg,
               (int)((td0 & 0xFFFFu) << 16),                      // tensor_dim0[15:0] (atomic addr 0)
               (int)((td0 >> 16) | ((td1 & 0xFFFFu) << 16)),      // td0[31:16] | td1[15:0]
               (int)((td1 >> 16) | (tile0 << 16)),                // td1[31:16] | tile_dim0
               (int)(tile1 & 0xFFFFu),                            // tile_dim1 | tile_dim2=0
               (int)stride0,                                      // tensor_dim0_stride[31:0]
               0, 0 };
    v4i z4 = {0, 0, 0, 0};
#if __clang_major__ >= 23
    v8i z8 = {0, 0, 0, 0, 0, 0, 0, 0};
    __builtin_amdgcn_tensor_load_to_lds(g0, g1, z4, z4, z8, 0);
#else
    __builtin_amdgcn_tensor_load_to_lds(g0, g1, z4, z4, 0);
#endif
}
#endif

__device__ __forceinline__ unsigned short f2bf(float x) {
    union { float f; unsigned int u; } c; c.f = x;
    unsigned int u = c.u;
    u += 0x7FFFu + ((u >> 16) & 1u);
    return (unsigned short)(u >> 16);
}

__device__ __forceinline__ float gelu_tanh(float x) {
    const float k0 = 0.7978845608028654f, k1 = 0.044715f;
    float x3 = x * x * x;
    return 0.5f * x * (1.0f + tanhf(k0 * (x + k1 * x3)));
}

__device__ __forceinline__ float wave_sum32(float v) {
    #pragma unroll
    for (int m = 16; m >= 1; m >>= 1) v += __shfl_xor(v, m, 32);
    return v;
}

// ---------------------------------------------------------------------------
// Prep: bf16 transposed weights. W1(256x512)->W1t(512x256), W2(512x256)->W2t(256x512)
__global__ __launch_bounds__(256) void prep_weights_k(
    const float* __restrict__ W1, const float* __restrict__ W2,
    unsigned short* __restrict__ W1t, unsigned short* __restrict__ W2t)
{
    int n = blockIdx.x, t = threadIdx.x;
    W1t[(size_t)n * 256 + t] = f2bf(W1[(size_t)t * 512 + n]);
    if (n < 256) {
        W2t[(size_t)n * 512 + t]       = f2bf(W2[(size_t)t * 256 + n]);
        W2t[(size_t)n * 512 + t + 256] = f2bf(W2[(size_t)(t + 256) * 256 + n]);
    }
}

// ---------------------------------------------------------------------------
// Prep: addvec[b,p,c] = (clip_p[b,p,:] @ Wv[p] + bv[p]) @ Wo[p] + bo[p]
__global__ __launch_bounds__(256) void prep_addvec_k(
    const float* __restrict__ clip, const float* __restrict__ Wv,
    const float* __restrict__ bv,   const float* __restrict__ Wo,
    const float* __restrict__ bo,   float* __restrict__ addvec)
{
    int bp = blockIdx.x, b = bp >> 4, p = bp & 15;
    __shared__ float csh[256];
    __shared__ float vsh[64];
    int t = threadIdx.x;
    csh[t] = clip[((size_t)b * 17 + p) * 256 + t];
    __syncthreads();
    if (t < 64) {
        const float* wv = Wv + (size_t)p * 256 * 64;
        float s = bv[p * 64 + t];
        for (int c = 0; c < 256; ++c) s += csh[c] * wv[c * 64 + t];
        vsh[t] = s;
    }
    __syncthreads();
    const float* wo = Wo + (size_t)p * 64 * 256;
    float s = bo[p * 256 + t];
    for (int d = 0; d < 64; ++d) s += vsh[d] * wo[d * 256 + t];
    addvec[(size_t)bp * 256 + t] = s;
}

// ---------------------------------------------------------------------------
// Prep: add2[b,c] = (kv[b,:] @ Wv2 + bv2) @ Wo2 + bo2 ; kv = clip[:, 16, :]
__global__ __launch_bounds__(256) void prep_add2_k(
    const float* __restrict__ clip, const float* __restrict__ Wv2,
    const float* __restrict__ bv2,  const float* __restrict__ Wo2,
    const float* __restrict__ bo2,  float* __restrict__ add2)
{
    int b = blockIdx.x;
    __shared__ float ksh[256];
    __shared__ float vsh[128];
    int t = threadIdx.x;
    ksh[t] = clip[((size_t)b * 17 + 16) * 256 + t];
    __syncthreads();
    if (t < 128) {
        float s = bv2[t];
        for (int c = 0; c < 256; ++c) s += ksh[c] * Wv2[c * 128 + t];
        vsh[t] = s;
    }
    __syncthreads();
    float s = bo2[t];
    for (int d = 0; d < 128; ++d) s += vsh[d] * Wo2[d * 256 + t];
    add2[b * 256 + t] = s;
}

// ---------------------------------------------------------------------------
// LN1: x[b,h*64+w,c] = image[b,c,h,w] + addvec[b,p(h,w),c] + add2[b,c]; layernorm.
// Image tile (256 c x 32 w, row stride 4096) DMA'd by TDM into LDS [c][w] (33-pad).
__global__ __launch_bounds__(256) void ln1_k(
    const float* __restrict__ img, const float* __restrict__ addvec,
    const float* __restrict__ add2, const float* __restrict__ g,
    const float* __restrict__ bb, float* __restrict__ xf,
    unsigned short* __restrict__ xbf)
{
    __shared__ __align__(16) float tile[256 * 33];
    int blk = blockIdx.x;
    int b = blk >> 7, rem = blk & 127;
    int h = rem >> 1, wb = (rem & 1) << 5;
    int t = threadIdx.x;
    int wave = t >> 5, lane = t & 31;
#if HAVE_TDM
    if (wave == 0) {
        tdm_load_2d((unsigned int)(uintptr_t)&tile[0],
                    (unsigned long long)(uintptr_t)(img + ((size_t)b * 256 * 64 + h) * 64 + wb),
                    TDW_F32, 32, 256, 32, 256, 4096);
        __builtin_amdgcn_s_wait_tensorcnt(0);
    }
    __syncthreads();
#else
    for (int k = 0; k < 32; ++k) {
        int c = k * 8 + (t >> 5);
        int wq = t & 31;
        tile[c * 33 + wq] = img[(((size_t)b * 256 + c) * 64 + h) * 64 + wb + wq];
    }
    __syncthreads();
#endif
    const float* a2 = add2 + b * 256;
    for (int rr = 0; rr < 4; ++rr) {
        int wq = wave * 4 + rr;
        int p = ((h >> 4) << 2) + ((wb + wq) >> 4);
        const float* av = addvec + ((size_t)b * 16 + p) * 256;
        float xv[8], s = 0.f, s2 = 0.f;
        #pragma unroll
        for (int k = 0; k < 8; ++k) {
            int c = k * 32 + lane;
            float x = tile[c * 33 + wq] + av[c] + a2[c];
            xv[k] = x; s += x; s2 += x * x;
        }
        s = wave_sum32(s); s2 = wave_sum32(s2);
        float mu  = s * (1.f / 256.f);
        float inv = rsqrtf(s2 * (1.f / 256.f) - mu * mu + 1e-5f);
        size_t R = ((size_t)b * 4096 + h * 64 + wb + wq) * 256;
        #pragma unroll
        for (int k = 0; k < 8; ++k) {
            int c = k * 32 + lane;
            float y = (xv[k] - mu) * inv * g[c] + bb[c];
            xf[R + c]  = y;
            xbf[R + c] = f2bf(y);
        }
    }
}

// ---------------------------------------------------------------------------
#define SA 40   // halfs per LDS row: 32 data + 8 pad -> 16B-aligned fragments, conflict-free

__device__ __forceinline__ void load_frag(Frag& f, const unsigned short* base, int row, int hi) {
    const unsigned short* p = base + row * SA + (hi << 3);
    f.q[0] = *(const v4u*)(p);
    f.q[1] = *(const v4u*)(p + 16);
}

// GEMM1: H = gelu(Xbf(65536x256) @ W1 + b1) -> bf16 (65536x512)
// block tile 128x128, K-chunks of 32 double-buffered via TDM; 8 waves, each 32x64
__global__ __launch_bounds__(256) void gemm1_k(
    const unsigned short* __restrict__ Abf, const unsigned short* __restrict__ Bt,
    const float* __restrict__ bias, unsigned short* __restrict__ Hbf)
{
    __shared__ __align__(16) unsigned short As[2][128 * SA];
    __shared__ __align__(16) unsigned short Bs[2][128 * SA];
    int t = threadIdx.x;
    int wave = t >> 5, lane = t & 31, nl = lane & 15, hi = lane >> 4;
    size_t mBase = (size_t)blockIdx.x * 128;
    int    nBase = blockIdx.y * 128;
    int waveM = wave >> 1, waveN = wave & 1;
    v8f acc[2][4];
    #pragma unroll
    for (int i = 0; i < 2; ++i)
        #pragma unroll
        for (int j = 0; j < 4; ++j) acc[i][j] = (v8f){0,0,0,0,0,0,0,0};

#if HAVE_TDM
    unsigned int ldsA[2] = { (unsigned int)(uintptr_t)&As[0][0], (unsigned int)(uintptr_t)&As[1][0] };
    unsigned int ldsB[2] = { (unsigned int)(uintptr_t)&Bs[0][0], (unsigned int)(uintptr_t)&Bs[1][0] };
    if (wave == 0)
        tdm_load_2d(ldsA[0], (unsigned long long)(uintptr_t)(Abf + mBase * 256),
                    TDW_BF16, 32, 128, 32, 128, 256);
    if (wave == 1)
        tdm_load_2d(ldsB[0], (unsigned long long)(uintptr_t)(Bt + (size_t)nBase * 256),
                    TDW_BF16, 32, 128, 32, 128, 256);
#endif
    int cur = 0;
    for (int kc = 0; kc < 256; kc += 32) {
#if HAVE_TDM
        if (wave < 2) __builtin_amdgcn_s_wait_tensorcnt(0);
        __syncthreads();
        if (kc + 32 < 256) {
            if (wave == 0)
                tdm_load_2d(ldsA[cur ^ 1], (unsigned long long)(uintptr_t)(Abf + mBase * 256 + kc + 32),
                            TDW_BF16, 32, 128, 32, 128, 256);
            if (wave == 1)
                tdm_load_2d(ldsB[cur ^ 1], (unsigned long long)(uintptr_t)(Bt + (size_t)nBase * 256 + kc + 32),
                            TDW_BF16, 32, 128, 32, 128, 256);
        }
#else
        #pragma unroll
        for (int r = 0; r < 8; ++r) {
            int pi = r * 256 + t;
            int m = pi >> 4, kd = pi & 15;
            *(unsigned int*)(&As[cur][0] + m * SA + kd * 2) =
                *(const unsigned int*)(Abf + (mBase + m) * 256 + kc + kd * 2);
            *(unsigned int*)(&Bs[cur][0] + m * SA + kd * 2) =
                *(const unsigned int*)(Bt + (size_t)(nBase + m) * 256 + kc + kd * 2);
        }
        __syncthreads();
#endif
        const unsigned short* Ab = &As[cur][0];
        const unsigned short* Bb = &Bs[cur][0];
        Frag a[2], bfr[4];
        #pragma unroll
        for (int i = 0; i < 2; ++i) load_frag(a[i], Ab, waveM * 32 + i * 16 + nl, hi);
        #pragma unroll
        for (int j = 0; j < 4; ++j) load_frag(bfr[j], Bb, waveN * 64 + j * 16 + nl, hi);
        #pragma unroll
        for (int i = 0; i < 2; ++i)
            #pragma unroll
            for (int j = 0; j < 4; ++j)
                acc[i][j] = __builtin_amdgcn_wmma_f32_16x16x32_bf16(
                    false, a[i].v, false, bfr[j].v, (short)0, acc[i][j], false, false);
#if !HAVE_TDM
        __syncthreads();
#endif
        cur ^= 1;
    }
    // epilogue: bias + gelu -> bf16
    #pragma unroll
    for (int i = 0; i < 2; ++i) {
        size_t mrow0 = mBase + waveM * 32 + i * 16 + hi * 8;
        #pragma unroll
        for (int j = 0; j < 4; ++j) {
            int n = nBase + waveN * 64 + j * 16 + nl;
            float bn = bias[n];
            #pragma unroll
            for (int r = 0; r < 8; ++r)
                Hbf[(mrow0 + r) * 512 + n] = f2bf(gelu_tanh(acc[i][j][r] + bn));
        }
    }
}

// ---------------------------------------------------------------------------
// GEMM2 + residual + LN2 + transpose store to out[b,c,h,w].
// block tile 32 x 256 (full N), K=512 chunks of 32 double-buffered via TDM.
struct G2Stage {
    unsigned short As[2][32 * SA];
    unsigned short Bs[2][256 * SA];
};
union G2Shared {
    G2Stage st;
    float tile[32 * 260];
};

__global__ __launch_bounds__(256) void gemm2_ln2_k(
    const unsigned short* __restrict__ Hbf, const unsigned short* __restrict__ W2t,
    const float* __restrict__ b2, const float* __restrict__ xres,
    const float* __restrict__ g2, const float* __restrict__ bln2,
    float* __restrict__ out)
{
    __shared__ __align__(16) G2Shared sh;
    int t = threadIdx.x;
    int wave = t >> 5, lane = t & 31, nl = lane & 15, hi = lane >> 4;
    size_t mBase = (size_t)blockIdx.x * 32;
    int mtile = wave & 1;
    int nb0   = (wave >> 1) * 4;
    v8f acc[4];
    #pragma unroll
    for (int j = 0; j < 4; ++j) acc[j] = (v8f){0,0,0,0,0,0,0,0};

#if HAVE_TDM
    unsigned int ldsA[2] = { (unsigned int)(uintptr_t)&sh.st.As[0][0], (unsigned int)(uintptr_t)&sh.st.As[1][0] };
    unsigned int ldsB[2] = { (unsigned int)(uintptr_t)&sh.st.Bs[0][0], (unsigned int)(uintptr_t)&sh.st.Bs[1][0] };
    if (wave == 0)
        tdm_load_2d(ldsA[0], (unsigned long long)(uintptr_t)(Hbf + mBase * 512),
                    TDW_BF16, 32, 32, 32, 32, 512);
    if (wave == 1)
        tdm_load_2d(ldsB[0], (unsigned long long)(uintptr_t)W2t,
                    TDW_BF16, 32, 256, 32, 256, 512);
#endif
    int cur = 0;
    for (int kc = 0; kc < 512; kc += 32) {
#if HAVE_TDM
        if (wave < 2) __builtin_amdgcn_s_wait_tensorcnt(0);
        __syncthreads();
        if (kc + 32 < 512) {
            if (wave == 0)
                tdm_load_2d(ldsA[cur ^ 1], (unsigned long long)(uintptr_t)(Hbf + mBase * 512 + kc + 32),
                            TDW_BF16, 32, 32, 32, 32, 512);
            if (wave == 1)
                tdm_load_2d(ldsB[cur ^ 1], (unsigned long long)(uintptr_t)(W2t + kc + 32),
                            TDW_BF16, 32, 256, 32, 256, 512);
        }
#else
        {
            int pi = t;                              // A: 32 rows x 16 dwords
            int m = pi >> 4, kd = pi & 15;
            if (t < 512) { }                          // 256 threads cover 512 dwords in 2 steps
            #pragma unroll
            for (int r = 0; r < 2; ++r) {
                int p2 = r * 256 + t;
                int mm = p2 >> 4, kk = p2 & 15;
                *(unsigned int*)(&sh.st.As[cur][0] + mm * SA + kk * 2) =
                    *(const unsigned int*)(Hbf + (mBase + mm) * 512 + kc + kk * 2);
            }
            #pragma unroll
            for (int r = 0; r < 16; ++r) {
                int p2 = r * 256 + t;
                int nn = p2 >> 4, kk = p2 & 15;
                *(unsigned int*)(&sh.st.Bs[cur][0] + nn * SA + kk * 2) =
                    *(const unsigned int*)(W2t + (size_t)nn * 512 + kc + kk * 2);
            }
            (void)m; (void)kd; (void)pi;
        }
        __syncthreads();
#endif
        const unsigned short* Ab = &sh.st.As[cur][0];
        const unsigned short* Bb = &sh.st.Bs[cur][0];
        Frag a;
        load_frag(a, Ab, mtile * 16 + nl, hi);
        #pragma unroll
        for (int j4 = 0; j4 < 4; ++j4) {
            Frag bfr;
            load_frag(bfr, Bb, (nb0 + j4) * 16 + nl, hi);
            acc[j4] = __builtin_amdgcn_wmma_f32_16x16x32_bf16(
                false, a.v, false, bfr.v, (short)0, acc[j4], false, false);
        }
#if !HAVE_TDM
        __syncthreads();
#endif
        cur ^= 1;
    }
    __syncthreads();   // staging no longer needed; tile aliases it
    // epilogue: + b2 + residual into LDS tile
    #pragma unroll
    for (int j4 = 0; j4 < 4; ++j4) {
        int n = (nb0 + j4) * 16 + nl;
        float bn = b2[n];
        #pragma unroll
        for (int r = 0; r < 8; ++r) {
            int m = mtile * 16 + hi * 8 + r;
            sh.tile[m * 260 + n] = acc[j4][r] + bn + xres[(mBase + m) * 256 + n];
        }
    }
    __syncthreads();
    // LN2 per row
    for (int rr = 0; rr < 4; ++rr) {
        int m = wave + rr * 8;
        float xv[8], s = 0.f, s2 = 0.f;
        #pragma unroll
        for (int k = 0; k < 8; ++k) {
            int c = k * 32 + lane;
            float x = sh.tile[m * 260 + c];
            xv[k] = x; s += x; s2 += x * x;
        }
        s = wave_sum32(s); s2 = wave_sum32(s2);
        float mu  = s * (1.f / 256.f);
        float inv = rsqrtf(s2 * (1.f / 256.f) - mu * mu + 1e-5f);
        #pragma unroll
        for (int k = 0; k < 8; ++k) {
            int c = k * 32 + lane;
            sh.tile[m * 260 + c] = (xv[k] - mu) * inv * g2[c] + bln2[c];
        }
    }
    __syncthreads();
    // coalesced transpose store: 32 rows = consecutive w at fixed (b,h)
    int b   = (int)(mBase >> 12);
    int hw0 = (int)(mBase & 4095);
    int h   = hw0 >> 6;
    int w0  = hw0 & 63;
    for (int k = 0; k < 32; ++k) {
        int c  = k * 8 + (t >> 5);
        int wl = t & 31;
        out[(((size_t)b * 256 + c) * 64 + h) * 64 + w0 + wl] = sh.tile[wl * 260 + c];
    }
}

// ---------------------------------------------------------------------------
extern "C" void kernel_launch(void* const* d_in, const int* in_sizes, int n_in,
                              void* d_out, int out_size, void* d_ws, size_t ws_size,
                              hipStream_t stream) {
    (void)in_sizes; (void)n_in; (void)out_size; (void)ws_size;
    const float* clip = (const float*)d_in[0];
    const float* img  = (const float*)d_in[1];
    // d_in[2..6], d_in[11..14] dead: softmax over singleton axis == 1
    const float* Wv   = (const float*)d_in[7];
    const float* bv   = (const float*)d_in[8];
    const float* Wo   = (const float*)d_in[9];
    const float* bo   = (const float*)d_in[10];
    const float* Wv2  = (const float*)d_in[15];
    const float* bv2  = (const float*)d_in[16];
    const float* Wo2  = (const float*)d_in[17];
    const float* bo2  = (const float*)d_in[18];
    const float* ln1g = (const float*)d_in[19];
    const float* ln1b = (const float*)d_in[20];
    const float* ln2g = (const float*)d_in[21];
    const float* ln2b = (const float*)d_in[22];
    const float* W1   = (const float*)d_in[23];
    const float* b1   = (const float*)d_in[24];
    const float* W2   = (const float*)d_in[25];
    const float* b2   = (const float*)d_in[26];
    float* out = (float*)d_out;

    char* wsb = (char*)d_ws;
    float*          xf     = (float*)(wsb + 0);                      // 64 MB
    unsigned short* xbf    = (unsigned short*)(wsb + 67108864);      // 32 MB
    unsigned short* hbf    = (unsigned short*)(wsb + 100663296);     // 64 MB
    unsigned short* w1t    = (unsigned short*)(wsb + 167772160);     // 256 KB
    unsigned short* w2t    = (unsigned short*)(wsb + 168034304);     // 256 KB
    float*          addvec = (float*)(wsb + 168296448);              // 256 KB
    float*          add2   = (float*)(wsb + 168558592);              // 16 KB

    prep_weights_k<<<512, 256, 0, stream>>>(W1, W2, w1t, w2t);
    prep_addvec_k <<<256, 256, 0, stream>>>(clip, Wv, bv, Wo, bo, addvec);
    prep_add2_k   <<<16,  256, 0, stream>>>(clip, Wv2, bv2, Wo2, bo2, add2);
    ln1_k         <<<2048, 256, 0, stream>>>(img, addvec, add2, ln1g, ln1b, xf, xbf);
    gemm1_k       <<<dim3(512, 4), 256, 0, stream>>>(xbf, w1t, b1, hbf);
    gemm2_ln2_k   <<<2048, 256, 0, stream>>>(hbf, w2t, b2, xf, ln2g, ln2b, out);
}
